// NAOMI_36885179138402
// MI455X (gfx1250) — compile-verified, matrix operand order checked by reading, not run on hardware
//
#include <hip/hip_runtime.h>
#include <hip/hip_bf16.h>

// ---------------------------------------------------------------------------
// Problem constants (from reference): T=256, B=128, Y=10, H=512, NL=2,
// HIGHEST=8, DEC=128.  3H = 1536.
// ---------------------------------------------------------------------------
#define CT  256
#define CB  128
#define CY  10
#define CH  512
#define CH3 1536
#define CDEC 128

typedef __attribute__((ext_vector_type(16))) __bf16         v16bf;
typedef __attribute__((ext_vector_type(8)))  float          v8f;
typedef __attribute__((ext_vector_type(16))) unsigned short v16us;
typedef __attribute__((ext_vector_type(8)))  unsigned short v8us;

__device__ __forceinline__ unsigned short f2bf(float f) {
    union { float f; unsigned int u; } v; v.f = f;
    unsigned int u = v.u;
    unsigned int lsb = (u >> 16) & 1u;
    u += 0x7fffu + lsb;                 // round-to-nearest-even
    return (unsigned short)(u >> 16);
}

__device__ __forceinline__ float sigm(float x) {
    return 1.0f / (1.0f + __expf(-x));
}

// Load a 16x32 bf16 WMMA operand fragment from a row-major matrix.
// Per CDNA5 ISA layout: lane L (half = L>>4, r = L&15) holds two contiguous
// 8-element chunks: cols [col0+8*half, +8) and [col0+16+8*half, +8).
// Works identically for A (rows = M) and B (rows = N of W, i.e. W row-major
// gives columns of W^T).
__device__ __forceinline__ v16bf load_frag(const unsigned short* base,
                                           int stride, int row0, int col0) {
    int lane = threadIdx.x & 31;
    int half = lane >> 4;
    int r    = row0 + (lane & 15);
    const unsigned short* p = base + (size_t)r * stride + col0 + half * 8;
    v8us lo = *(const v8us*)(p);
    v8us hi = *(const v8us*)(p + 16);
    v16us w = __builtin_shufflevector(lo, hi,
        0, 1, 2, 3, 4, 5, 6, 7, 8, 9, 10, 11, 12, 13, 14, 15);
    return __builtin_bit_cast(v16bf, w);
}

__device__ __forceinline__ v8f wmma_bf16(v8f c, v16bf a, v16bf b) {
    return __builtin_amdgcn_wmma_f32_16x16x32_bf16(
        /*neg_a=*/false, a, /*neg_b=*/false, b,
        /*c_mod=*/(short)0, c, /*reuse_a=*/false, /*reuse_b=*/false);
}

// ---------------------------------------------------------------------------
// Zero-init (hidden states, pres_rev buffer)
// ---------------------------------------------------------------------------
__global__ void k_zero(unsigned int* a, size_t na,
                       unsigned int* b, size_t nb,
                       unsigned int* c, size_t nc) {
    size_t i  = (size_t)blockIdx.x * blockDim.x + threadIdx.x;
    size_t st = (size_t)gridDim.x * blockDim.x;
    for (size_t x = i; x < na; x += st) a[x] = 0u;
    for (size_t x = i; x < nb; x += st) b[x] = 0u;
    for (size_t x = i; x < nc; x += st) c[x] = 0u;
}

// ---------------------------------------------------------------------------
// fp32 -> bf16 weight conversion
// ---------------------------------------------------------------------------
struct ConvPtrs {
    const float* w[6];    // 1536x512 recurrent matrices
    const float* dw[4];   // dec_W 128x1024
    const float* mw[4];   // mean_W 10x128 (padded to 16x128)
};

#define NWEL  786432u     // 1536*512
#define WTOT  (6u * NWEL)
#define DTOT  (4u * 131072u)
#define MTOT  (4u * 2048u)

__global__ void k_convert(ConvPtrs cp, unsigned short* wdst,
                          unsigned short* ddst, unsigned short* mdst) {
    size_t i  = (size_t)blockIdx.x * blockDim.x + threadIdx.x;
    size_t st = (size_t)gridDim.x * blockDim.x;
    size_t total = (size_t)WTOT + DTOT + MTOT;
    for (size_t x = i; x < total; x += st) {
        if (x < WTOT) {
            int m = (int)(x / NWEL);
            size_t o = x % NWEL;
            wdst[x] = f2bf(cp.w[m][o]);
        } else if (x < WTOT + DTOT) {
            size_t j = x - WTOT;
            int s = (int)(j / 131072u);
            size_t o = j % 131072u;
            ddst[j] = f2bf(cp.dw[s][o]);
        } else {
            size_t j = x - WTOT - DTOT;
            int s = (int)(j / 2048u);
            size_t o = j % 2048u;
            int n = (int)(o >> 7);
            int k = (int)(o & 127);
            mdst[j] = (n < CY) ? f2bf(cp.mw[s][n * CDEC + k]) : (unsigned short)0;
        }
    }
}

// ---------------------------------------------------------------------------
// GRU layer 0 (one time step).  Blocks 0..7: fwd scan, blocks 8..15: bwd.
// Each block owns a 16-row batch tile.  The wave's full K=512 A-strip
// (16 fragments = 128 VGPRs) is hoisted into registers once; the column-tile
// loop then streams only B fragments from global.  __launch_bounds__(256,1)
// raises the VGPR budget (only 16 blocks exist -> occupancy is irrelevant)
// so the strip stays resident with no scratch spills.
// Dynamic LDS: 16*512 bf16 + 16*1536 f32.
// ---------------------------------------------------------------------------
__global__ __launch_bounds__(256, 1)
void k_l0(int t,
          const float* gt, const float* data,
          const float* WihF, const float* bihF, const float* bhhF,
          const unsigned short* WhhF, float* h0F, unsigned short* h0Fb,
          const float* WihB, const float* bihB, const float* bhhB,
          const unsigned short* WhhB, float* h0B, unsigned short* h0Bb) {
    extern __shared__ char smem[];
    unsigned short* Ah = (unsigned short*)smem;            // 16 x 512 bf16
    float*          GH = (float*)(smem + 16 * CH * 2);     // 16 x 1536 f32

    const bool bwd = (blockIdx.x >= 8);
    if (bwd && t >= CT - 1) return;                        // bwd has 255 steps
    const int r0  = (blockIdx.x & 7) * 16;
    const int tid = threadIdx.x;

    const float* Wih; const float* bih; const float* bhh;
    const unsigned short* Whh; float* hF; unsigned short* hBf;
    const float* xrow; int xin;
    if (!bwd) {
        Wih = WihF; bih = bihF; bhh = bhhF; Whh = WhhF; hF = h0F; hBf = h0Fb;
        xrow = gt + (size_t)t * CB * CY;           xin = CY;
    } else {
        Wih = WihB; bih = bihB; bhh = bhhB; Whh = WhhB; hF = h0B; hBf = h0Bb;
        xrow = data + (size_t)(CT - 1 - t) * CB * (CY + 1); xin = CY + 1;
    }

    // Stage A tile (this block's 16 rows of h, bf16) into LDS.
    for (int i = tid; i < (16 * CH) / 8; i += 256) {
        int row = i >> 6;
        int col = (i & 63) * 8;
        *(v8us*)(Ah + row * CH + col) =
            *(const v8us*)(hBf + (size_t)(r0 + row) * CH + col);
    }
    __syncthreads();

    const int wave = tid >> 5;
    const int lane = tid & 31;

    // Hoist the full A strip into registers (16 x v16bf = 128 VGPRs).
    v16bf afr[CH / 32];
#pragma unroll
    for (int kk = 0; kk < CH / 32; ++kk)
        afr[kk] = load_frag(Ah, CH, 0, kk * 32);

    for (int ct = wave; ct < CH3 / 16; ct += 8) {
        int n0 = ct * 16;
        v8f acc = {0.f, 0.f, 0.f, 0.f, 0.f, 0.f, 0.f, 0.f};
#pragma unroll
        for (int kk = 0; kk < CH / 32; ++kk) {
            v16bf b = load_frag(Whh, CH, n0, kk * 32);
            acc = wmma_bf16(acc, afr[kk], b);
        }
        int half = lane >> 4, n = lane & 15;
#pragma unroll
        for (int v = 0; v < 8; ++v)
            GH[(v + 8 * half) * CH3 + n0 + n] = acc[v];
    }
    __syncthreads();

    for (int e = tid; e < 16 * CH; e += 256) {
        int row = e >> 9, j = e & (CH - 1);
        int b = r0 + row;
        float ghr = GH[row * CH3 + j]            + bhh[j];
        float ghz = GH[row * CH3 + CH + j]       + bhh[CH + j];
        float ghn = GH[row * CH3 + 2 * CH + j]   + bhh[2 * CH + j];
        float gir = bih[j], giz = bih[CH + j], gin = bih[2 * CH + j];
        const float* x = xrow + (size_t)b * xin;
        for (int k = 0; k < xin; ++k) {
            float xv = x[k];
            gir += xv * Wih[(size_t)j * xin + k];
            giz += xv * Wih[(size_t)(CH + j) * xin + k];
            gin += xv * Wih[(size_t)(2 * CH + j) * xin + k];
        }
        float r  = sigm(gir + ghr);
        float z  = sigm(giz + ghz);
        float nn = tanhf(gin + r * ghn);
        float hold = hF[(size_t)b * CH + j];
        float hnew = (1.0f - z) * nn + z * hold;
        hF [(size_t)b * CH + j] = hnew;
        hBf[(size_t)b * CH + j] = f2bf(hnew);
    }
}

// ---------------------------------------------------------------------------
// GRU layer 1 (one time step).  Two passes, each with a register-resident
// A strip: pass 0 computes gi = h0_new @ Wih1^T into GI, pass 1 computes
// gh = h1 @ Whh1^T into GH.  Then gate math, h1 update, scan-output store.
// Dynamic LDS: 2*(16*512 bf16) + 2*(16*1536 f32) = 224 KB.
// ---------------------------------------------------------------------------
__global__ __launch_bounds__(256, 1)
void k_l1(int t,
          const float* bihF, const float* bhhF,
          const unsigned short* WihF, const unsigned short* WhhF,
          const unsigned short* h0Fb, float* h1F, unsigned short* h1Fb,
          unsigned short* hsOut,
          const float* bihB, const float* bhhB,
          const unsigned short* WihB, const unsigned short* WhhB,
          const unsigned short* h0Bb, float* h1B, unsigned short* h1Bb,
          unsigned short* presOut) {
    extern __shared__ char smem[];
    unsigned short* A0 = (unsigned short*)smem;                 // h0 tile
    unsigned short* A1 = (unsigned short*)(smem + 16 * CH * 2); // h1 tile
    float* GI = (float*)(smem + 2 * 16 * CH * 2);
    float* GH = GI + 16 * CH3;

    const bool bwd = (blockIdx.x >= 8);
    if (bwd && t >= CT - 1) return;
    const int r0  = (blockIdx.x & 7) * 16;
    const int tid = threadIdx.x;

    const float* bih; const float* bhh;
    const unsigned short *Wih, *Whh, *h0b;
    float* hF; unsigned short* hBf;
    if (!bwd) { bih = bihF; bhh = bhhF; Wih = WihF; Whh = WhhF;
                h0b = h0Fb; hF = h1F; hBf = h1Fb; }
    else      { bih = bihB; bhh = bhhB; Wih = WihB; Whh = WhhB;
                h0b = h0Bb; hF = h1B; hBf = h1Bb; }

    for (int i = tid; i < (16 * CH) / 8; i += 256) {
        int row = i >> 6;
        int col = (i & 63) * 8;
        *(v8us*)(A0 + row * CH + col) =
            *(const v8us*)(h0b + (size_t)(r0 + row) * CH + col);
        *(v8us*)(A1 + row * CH + col) =
            *(const v8us*)(hBf + (size_t)(r0 + row) * CH + col);
    }
    __syncthreads();

    const int wave = tid >> 5;
    const int lane = tid & 31;

    for (int pass = 0; pass < 2; ++pass) {
        const unsigned short* Asrc = pass ? A1 : A0;
        const unsigned short* Wsrc = pass ? Whh : Wih;
        float* Gdst = pass ? GH : GI;

        v16bf afr[CH / 32];
#pragma unroll
        for (int kk = 0; kk < CH / 32; ++kk)
            afr[kk] = load_frag(Asrc, CH, 0, kk * 32);

        for (int ct = wave; ct < CH3 / 16; ct += 8) {
            int n0 = ct * 16;
            v8f acc = {0.f, 0.f, 0.f, 0.f, 0.f, 0.f, 0.f, 0.f};
#pragma unroll
            for (int kk = 0; kk < CH / 32; ++kk) {
                v16bf b = load_frag(Wsrc, CH, n0, kk * 32);
                acc = wmma_bf16(acc, afr[kk], b);
            }
            int half = lane >> 4, n = lane & 15;
#pragma unroll
            for (int v = 0; v < 8; ++v)
                Gdst[(v + 8 * half) * CH3 + n0 + n] = acc[v];
        }
    }
    __syncthreads();

    for (int e = tid; e < 16 * CH; e += 256) {
        int row = e >> 9, j = e & (CH - 1);
        int b = r0 + row;
        float gir = GI[row * CH3 + j]          + bih[j];
        float giz = GI[row * CH3 + CH + j]     + bih[CH + j];
        float gin = GI[row * CH3 + 2 * CH + j] + bih[2 * CH + j];
        float ghr = GH[row * CH3 + j]          + bhh[j];
        float ghz = GH[row * CH3 + CH + j]     + bhh[CH + j];
        float ghn = GH[row * CH3 + 2 * CH + j] + bhh[2 * CH + j];
        float r  = sigm(gir + ghr);
        float z  = sigm(giz + ghz);
        float nn = tanhf(gin + r * ghn);
        float hold = hF[(size_t)b * CH + j];
        float hnew = (1.0f - z) * nn + z * hold;
        hF [(size_t)b * CH + j] = hnew;
        unsigned short hb = f2bf(hnew);
        hBf[(size_t)b * CH + j] = hb;
        if (!bwd) {
            hsOut[((size_t)t * CB + b) * CH + j] = hb;          // hs[t]
        } else if (t <= CT - 3) {                               // pres_rev[253-t]
            presOut[((size_t)(CT - 3 - t) * CB + b) * CH + j] = hb;
        }
    }
}

// ---------------------------------------------------------------------------
// Decoder + loss.  One wave per (scale, t, 16-row batch tile); 998 blocks x 8
// waves == 7984 tasks exactly.  dec = relu(hcat @ dec_W^T + dec_b) with each
// 512-wide half of the hcat A strip hoisted into registers and 8 persistent
// N-tile accumulators; LDS transpose; mean head via WMMA (K=128, N padded to
// 16); squared-error partials reduced per block (deterministic order).
// ---------------------------------------------------------------------------
__global__ __launch_bounds__(256, 1)
void k_dec(const unsigned short* hs, const unsigned short* pres,
           const unsigned short* decW, const unsigned short* meanW,
           const float* db0, const float* mb0,
           const float* db1, const float* mb1,
           const float* db2, const float* mb2,
           const float* db3, const float* mb3,
           const float* gt, float* partials) {
    __shared__ unsigned short decS[8 * 16 * CDEC];   // 32 KB
    __shared__ float red[256];

    const int tid  = threadIdx.x;
    const int wave = tid >> 5;
    const int lane = tid & 31;
    const int gw   = blockIdx.x * 8 + wave;

    int s, base;
    if      (gw < 2040) { s = 0; base = 0;    }
    else if (gw < 4064) { s = 1; base = 2040; }
    else if (gw < 6056) { s = 2; base = 4064; }
    else                { s = 3; base = 6056; }
    const int local = gw - base;
    const int t     = local >> 3;
    const int b0    = (local & 7) * 16;
    const int step  = 1 << s;

    const float* db = (s == 0) ? db0 : (s == 1) ? db1 : (s == 2) ? db2 : db3;
    const float* mb = (s == 0) ? mb0 : (s == 1) ? mb1 : (s == 2) ? mb2 : mb3;
    const unsigned short* dW = decW  + (size_t)s * (CDEC * 2 * CH);
    const unsigned short* mW = meanW + (size_t)s * (16 * CDEC);
    const unsigned short* hsrow = hs + ((size_t)t * CB + b0) * CH;
    const int u = 2 * step - 2 + t;
    const unsigned short* prow = pres + ((size_t)u * CB + b0) * CH;
    unsigned short* myDec = decS + wave * 16 * CDEC;

    v8f accs[CDEC / 16];
#pragma unroll
    for (int ct = 0; ct < CDEC / 16; ++ct)
        accs[ct] = (v8f){0.f, 0.f, 0.f, 0.f, 0.f, 0.f, 0.f, 0.f};

    for (int halfk = 0; halfk < 2; ++halfk) {
        const unsigned short* arow = halfk ? prow : hsrow;
        const int kbase = halfk * CH;

        v16bf afr[CH / 32];
#pragma unroll
        for (int kk = 0; kk < CH / 32; ++kk)
            afr[kk] = load_frag(arow, CH, 0, kk * 32);

#pragma unroll
        for (int ct = 0; ct < CDEC / 16; ++ct) {
#pragma unroll
            for (int kk = 0; kk < CH / 32; ++kk) {
                v16bf b = load_frag(dW, 2 * CH, ct * 16, kbase + kk * 32);
                accs[ct] = wmma_bf16(accs[ct], afr[kk], b);
            }
        }
    }

    {
        int half = lane >> 4, n = lane & 15;
#pragma unroll
        for (int ct = 0; ct < CDEC / 16; ++ct) {
            int n0 = ct * 16;
            float bias = db[n0 + n];
#pragma unroll
            for (int v = 0; v < 8; ++v) {
                float val = accs[ct][v] + bias;
                val = (val > 0.f) ? val : 0.f;                 // ReLU
                myDec[(v + 8 * half) * CDEC + n0 + n] = f2bf(val);
            }
        }
    }
    __syncthreads();

    v8f macc = {0.f, 0.f, 0.f, 0.f, 0.f, 0.f, 0.f, 0.f};
#pragma unroll
    for (int kk = 0; kk < CDEC / 32; ++kk) {
        v16bf a = load_frag(myDec, CDEC, 0, kk * 32);
        v16bf b = load_frag(mW, CDEC, 0, kk * 32);
        macc = wmma_bf16(macc, a, b);
    }

    int half = lane >> 4, n = lane & 15;
    float lsum = 0.f;
    if (n < CY) {
        float bias = mb[n];
        const float* tg = gt + ((size_t)(step + t) * CB + b0) * CY;
#pragma unroll
        for (int v = 0; v < 8; ++v) {
            int row = v + 8 * half;
            float m = macc[v] + bias;
            float d = m - tg[(size_t)row * CY + n];
            lsum += d * d;
        }
    }
    red[tid] = lsum;
    __syncthreads();
    for (int off = 128; off > 0; off >>= 1) {
        if (tid < off) red[tid] += red[tid + off];
        __syncthreads();
    }
    if (tid == 0) partials[blockIdx.x] = red[0];
}

__global__ void k_final(const float* partials, int n, float* out) {
    __shared__ float red[256];
    int tid = threadIdx.x;
    float s = 0.f;
    for (int i = tid; i < n; i += 256) s += partials[i];
    red[tid] = s;
    __syncthreads();
    for (int off = 128; off > 0; off >>= 1) {
        if (tid < off) red[tid] += red[tid + off];
        __syncthreads();
    }
    if (tid == 0) out[0] = red[0] * (1.0f / ((float)CT * (float)CB));
}

// ---------------------------------------------------------------------------
// Host driver (graph-capture safe: launches only)
// ---------------------------------------------------------------------------
extern "C" void kernel_launch(void* const* d_in, const int* in_sizes, int n_in,
                              void* d_out, int out_size, void* d_ws, size_t ws_size,
                              hipStream_t stream) {
    (void)in_sizes; (void)n_in; (void)out_size; (void)ws_size;

    const float* data  = (const float*)d_in[0];
    const float* gt    = (const float*)d_in[1];
    const float* fWih0 = (const float*)d_in[2];
    const float* fWih1 = (const float*)d_in[3];
    const float* fWhh0 = (const float*)d_in[4];
    const float* fWhh1 = (const float*)d_in[5];
    const float* fbih0 = (const float*)d_in[6];
    const float* fbih1 = (const float*)d_in[7];
    const float* fbhh0 = (const float*)d_in[8];
    const float* fbhh1 = (const float*)d_in[9];
    const float* bWih0 = (const float*)d_in[10];
    const float* bWih1 = (const float*)d_in[11];
    const float* bWhh0 = (const float*)d_in[12];
    const float* bWhh1 = (const float*)d_in[13];
    const float* bbih0 = (const float*)d_in[14];
    const float* bbih1 = (const float*)d_in[15];
    const float* bbhh0 = (const float*)d_in[16];
    const float* bbhh1 = (const float*)d_in[17];
    const float* dWs[4], *dbs[4], *mWs[4], *mbs[4];
    for (int s = 0; s < 4; ++s) {
        dWs[s] = (const float*)d_in[18 + 4 * s];
        dbs[s] = (const float*)d_in[19 + 4 * s];
        mWs[s] = (const float*)d_in[20 + 4 * s];
        mbs[s] = (const float*)d_in[21 + 4 * s];
    }

    // ---- workspace carve (bf16 region, then f32 region) ----
    unsigned short* wsu = (unsigned short*)d_ws;
    size_t o = 0;
    unsigned short* wAll  = wsu + o; o += (size_t)6 * NWEL;       // 6 recurrent mats
    unsigned short* decWb = wsu + o; o += (size_t)4 * 131072;     // dec_W bf16
    unsigned short* mWb   = wsu + o; o += (size_t)4 * 2048;       // mean_W padded
    unsigned short* hsB   = wsu + o; o += (size_t)CT * CB * CH;   // hs bf16
    unsigned short* presB = wsu + o; o += (size_t)(CT - 1) * CB * CH; // pres_rev
    unsigned short* h0Fb  = wsu + o; o += (size_t)CB * CH;
    unsigned short* h1Fb  = wsu + o; o += (size_t)CB * CH;
    unsigned short* h0Bb  = wsu + o; o += (size_t)CB * CH;
    unsigned short* h1Bb  = wsu + o; o += (size_t)CB * CH;
    float* wsf = (float*)(wsu + o);
    float* h0F = wsf;
    float* h1F = wsf + 1 * (size_t)CB * CH;
    float* h0B = wsf + 2 * (size_t)CB * CH;
    float* h1B = wsf + 3 * (size_t)CB * CH;
    float* partials = wsf + 4 * (size_t)CB * CH;                  // 998 floats

    const unsigned short* WhhF0b = wAll + 0 * (size_t)NWEL;
    const unsigned short* WihF1b = wAll + 1 * (size_t)NWEL;
    const unsigned short* WhhF1b = wAll + 2 * (size_t)NWEL;
    const unsigned short* WhhB0b = wAll + 3 * (size_t)NWEL;
    const unsigned short* WihB1b = wAll + 4 * (size_t)NWEL;
    const unsigned short* WhhB1b = wAll + 5 * (size_t)NWEL;

    // ---- init: zero pres_rev + hidden states ----
    k_zero<<<2048, 256, 0, stream>>>(
        (unsigned int*)presB, (size_t)(CT - 1) * CB * CH / 2,
        (unsigned int*)h0Fb,  (size_t)4 * CB * CH / 2,
        (unsigned int*)h0F,   (size_t)4 * CB * CH);

    // ---- convert weights ----
    ConvPtrs cp;
    cp.w[0] = fWhh0; cp.w[1] = fWih1; cp.w[2] = fWhh1;
    cp.w[3] = bWhh0; cp.w[4] = bWih1; cp.w[5] = bWhh1;
    for (int s = 0; s < 4; ++s) { cp.dw[s] = dWs[s]; cp.mw[s] = mWs[s]; }
    k_convert<<<2048, 256, 0, stream>>>(cp, wAll, decWb, mWb);

    // ---- recurrent scans (fwd + bwd fused per launch) ----
    const size_t sh0 = (size_t)16 * CH * 2 + (size_t)16 * CH3 * 4;       // 112 KB
    const size_t sh1 = 2 * (size_t)16 * CH * 2 + 2 * (size_t)16 * CH3 * 4; // 224 KB
    for (int t = 0; t < CT; ++t) {
        k_l0<<<16, 256, sh0, stream>>>(t, gt, data,
            fWih0, fbih0, fbhh0, WhhF0b, h0F, h0Fb,
            bWih0, bbih0, bbhh0, WhhB0b, h0B, h0Bb);
        k_l1<<<16, 256, sh1, stream>>>(t,
            fbih1, fbhh1, WihF1b, WhhF1b, h0Fb, h1F, h1Fb, hsB,
            bbih1, bbhh1, WihB1b, WhhB1b, h0Bb, h1B, h1Bb, presB);
    }

    // ---- decoders + loss ----
    k_dec<<<998, 256, 0, stream>>>(hsB, presB, decWb, mWb,
        dbs[0], mbs[0], dbs[1], mbs[1], dbs[2], mbs[2], dbs[3], mbs[3],
        gt, partials);
    k_final<<<1, 256, 0, stream>>>(partials, 998, (float*)d_out);
}
